// MaskedRNN_21818433863703
// MI455X (gfx1250) — matrix operand order repaired and checked by reading, not compiled, not run on hardware
//
#include <hip/hip_runtime.h>

typedef __attribute__((ext_vector_type(16))) _Float16 v16h;
typedef __attribute__((ext_vector_type(8)))  _Float16 v8h;
typedef __attribute__((ext_vector_type(8)))  float    v8f;
typedef __attribute__((ext_vector_type(4)))  float    v4f;
typedef int vsi4 __attribute__((vector_size(16)));   // matches builtin param type

#define T_LEN 512
#define B_SZ  64
#define H_SZ  256
#define G_SZ  768   // 3*H

#if __has_builtin(__builtin_amdgcn_global_load_async_to_lds_b128) && \
    __has_builtin(__builtin_amdgcn_s_wait_asynccnt)
#define USE_ASYNC_LDS 1
#else
#define USE_ASYNC_LDS 0
#endif

// ---------------------------------------------------------------------------
// Transpose + convert weight W [768 x 256] f32 (row-major, [g][e]) into
// Wt [256 x 768] f16 (row-major, [k][n]) so each lane's WMMA B-fragment
// (lane = N, 16 halves = K at stride 768 halves) takes 16 strided loads.
// ---------------------------------------------------------------------------
__global__ void wtrans_kernel(const float* __restrict__ W, _Float16* __restrict__ Wt) {
  int idx = blockIdx.x * 256 + threadIdx.x;            // over 768*256
  if (idx < G_SZ * H_SZ) {
    int n = idx >> 8;                                  // row of W (gate dim)
    int k = idx & 255;                                 // col of W (E/H dim)
    Wt[k * G_SZ + n] = (_Float16)W[idx];
  }
}

// ---------------------------------------------------------------------------
// x_proj[t*B+b][g] = sum_e emb[input[b][t]][e] * W_ih[g][e] + b_ih[g]
// M = 32768, N = 768, K = 256.  One 16x16 tile per wave,
// 8 x v_wmma_f32_16x16x32_f16 along K.  A converted f32->f16 in-register.
// ---------------------------------------------------------------------------
__global__ void __launch_bounds__(256) xproj_kernel(
    const int* __restrict__ inp, const float* __restrict__ emb,
    const _Float16* __restrict__ wihT, const float* __restrict__ b_ih,
    float* __restrict__ xproj) {
  const int lane  = threadIdx.x & 31;
  const int wave  = threadIdx.x >> 5;
  const int mtile = blockIdx.x;                        // 0..2047
  const int n0    = (blockIdx.y * 8 + wave) * 16;      // 0..752

  // A-fragment geometry (16-bit A 16x32, ISA 7.12.2)
  const int m    = lane & 15;
  const int koff = (lane >> 4) * 8;                    // lanes 16-31 take K+8 / K+24
  const int row  = mtile * 16 + m;                     // row of x_proj = t*64 + b
  const int t    = row >> 6;
  const int b    = row & 63;
  const long token = (long)inp[b * T_LEN + t];
  const float* arow = emb + token * H_SZ;

  v8f acc = {};
  #pragma unroll
  for (int kk = 0; kk < 8; ++kk) {
    const int k0 = kk * 32;
    const float* p0 = arow + k0 + koff;
    const float* p1 = p0 + 16;
    v16h a;
    #pragma unroll
    for (int i = 0; i < 8; ++i) { a[i] = (_Float16)p0[i]; a[8 + i] = (_Float16)p1[i]; }
    v16h bf;
    #pragma unroll
    for (int i = 0; i < 8; ++i) { bf[i] = wihT[(size_t)(k0 + koff + i) * G_SZ + n0 + m]; bf[8 + i] = wihT[(size_t)(k0 + koff + 16 + i) * G_SZ + n0 + m]; }
    acc = __builtin_amdgcn_wmma_f32_16x16x32_f16(false, a, false, bf,
                                                 (short)0, acc, false, false);
  }

  // C/D layout: VGPR i -> row i (+8 for lanes 16-31), col = lane&15
  const int   col  = lane & 15;
  const float bias = b_ih[n0 + col];
  const int   rhi  = (lane >> 4) * 8;
  #pragma unroll
  for (int i = 0; i < 8; ++i)
    xproj[(size_t)(mtile * 16 + rhi + i) * G_SZ + n0 + col] = acc[i] + bias;
}

// ---------------------------------------------------------------------------
// Sequential GRU scan. Batch rows are independent -> 4 workgroups, each owns
// a 16-row batch tile for all 512 steps.  16 waves/WG; wave w owns N-tiles
// 3w..3w+2 and keeps its 24 W_hh B-fragments in registers for the whole scan
// (192 VGPRs/lane) -- zero weight traffic inside the serial loop.
// The x_proj tile for step t is copied global->LDS asynchronously (ASYNCcnt)
// while the WMMAs run, removing the L2 round-trip from the critical path.
// ---------------------------------------------------------------------------
__global__ void __launch_bounds__(512) scan_kernel(
    const float* __restrict__ xproj, const _Float16* __restrict__ whhT,
    const float* __restrict__ b_hh, float* __restrict__ out) {
  __shared__ __align__(32) _Float16 hS[16 * H_SZ];     // 8 KB
  __shared__ __align__(32) float    hpS[16 * G_SZ];    // 48 KB
  __shared__ __align__(32) float    xS[16 * G_SZ];     // 48 KB
  const int lane = threadIdx.x & 31;
  const int wave = threadIdx.x >> 5;                   // 0..15
  const int bt   = blockIdx.x;                         // batch tile 0..3

  for (int i = threadIdx.x; i < 16 * H_SZ; i += 512) hS[i] = (_Float16)0.0f;

  // Hoist all W_hh fragments into registers (loop-invariant weights).
  v16h bfr[3][8];
  #pragma unroll
  for (int tt = 0; tt < 3; ++tt) {
    const int n0 = (wave * 3 + tt) * 16;
    #pragma unroll
    for (int kk = 0; kk < 8; ++kk)
      for (int i = 0; i < 8; ++i) { bfr[tt][kk][i] = whhT[(size_t)(kk * 32 + (lane >> 4) * 8 + i) * G_SZ + n0 + (lane & 15)]; bfr[tt][kk][8 + i] = whhT[(size_t)(kk * 32 + (lane >> 4) * 8 + 16 + i) * G_SZ + n0 + (lane & 15)]; }
  }
  __syncthreads();

  const int m    = lane & 15;
  const int koff = (lane >> 4) * 8;
  const int col  = lane & 15;
  const int rhi  = (lane >> 4) * 8;

  for (int t = 0; t < T_LEN; ++t) {
    // -- kick off the x_proj tile fetch for this step (overlaps the GEMM) --
    float* xrow = const_cast<float*>(xproj) + ((size_t)t * B_SZ + bt * 16) * G_SZ;
#if USE_ASYNC_LDS
    #pragma unroll
    for (int i = 0; i < 6; ++i) {
      __attribute__((address_space(1))) void* gsrc =
          (__attribute__((address_space(1))) void*)(xrow + threadIdx.x * 4 + i * 2048);
      __attribute__((address_space(3))) void* ldst =
          (__attribute__((address_space(3))) void*)(xS + threadIdx.x * 4 + i * 2048);
      __builtin_amdgcn_global_load_async_to_lds_b128(
          (__attribute__((address_space(1))) vsi4*)gsrc,
          (__attribute__((address_space(3))) vsi4*)ldst, 0, 0);
    }
#else
    v4f xpre[6];
    #pragma unroll
    for (int i = 0; i < 6; ++i)
      xpre[i] = *(const v4f*)(xrow + threadIdx.x * 4 + i * 2048);
#endif

    // -- hp = h @ W_hh^T (+ b_hh), staged to LDS --
    #pragma unroll
    for (int tt = 0; tt < 3; ++tt) {
      const int n0 = (wave * 3 + tt) * 16;
      v8f acc = {};
      #pragma unroll
      for (int kk = 0; kk < 8; ++kk) {
        const _Float16* hp = hS + m * H_SZ + kk * 32 + koff;
        v8h lo = *(const v8h*)hp;
        v8h hi = *(const v8h*)(hp + 16);
        v16h a;
        #pragma unroll
        for (int i = 0; i < 8; ++i) { a[i] = lo[i]; a[8 + i] = hi[i]; }
        acc = __builtin_amdgcn_wmma_f32_16x16x32_f16(false, a, false, bfr[tt][kk],
                                                     (short)0, acc, false, false);
      }
      const float bias = b_hh[n0 + col];
      #pragma unroll
      for (int i = 0; i < 8; ++i)
        hpS[(rhi + i) * G_SZ + n0 + col] = acc[i] + bias;
    }

#if USE_ASYNC_LDS
    __builtin_amdgcn_s_wait_asynccnt(0);
#else
    #pragma unroll
    for (int i = 0; i < 6; ++i)
      *(v4f*)(xS + threadIdx.x * 4 + i * 2048) = xpre[i];
#endif
    __syncthreads();

    // -- gate math: each thread owns distinct (b,j) elements of the h tile --
    for (int e = threadIdx.x; e < 16 * H_SZ; e += 512) {
      const int br = e >> 8;
      const int j  = e & 255;
      const float xr = xS[br * G_SZ + j];
      const float xz = xS[br * G_SZ + H_SZ + j];
      const float xn = xS[br * G_SZ + 2 * H_SZ + j];
      const float hr = hpS[br * G_SZ + j];
      const float hz = hpS[br * G_SZ + H_SZ + j];
      const float hn = hpS[br * G_SZ + 2 * H_SZ + j];
      const float r  = 1.0f / (1.0f + __expf(-(xr + hr)));
      const float z  = 1.0f / (1.0f + __expf(-(xz + hz)));
      const float n  = tanhf(xn + r * hn);
      const float hprev = (float)hS[br * H_SZ + j];
      const float hnew  = (1.0f - z) * n + z * hprev;
      out[((size_t)t * B_SZ + bt * 16 + br) * H_SZ + j] = hnew;
      hS[br * H_SZ + j] = (_Float16)hnew;
    }
    __syncthreads();
  }

  // final hidden state, appended after outputs
  for (int e = threadIdx.x; e < 16 * H_SZ; e += 512) {
    const int br = e >> 8, j = e & 255;
    out[(size_t)T_LEN * B_SZ * H_SZ + (bt * 16 + br) * H_SZ + j] = (float)hS[br * H_SZ + j];
  }
}

// ---------------------------------------------------------------------------
extern "C" void kernel_launch(void* const* d_in, const int* in_sizes, int n_in,
                              void* d_out, int out_size, void* d_ws, size_t ws_size,
                              hipStream_t stream) {
  const int*   inp = (const int*)d_in[0];
  const float* emb = (const float*)d_in[1];
  const float* Wih = (const float*)d_in[2];
  const float* Whh = (const float*)d_in[3];
  const float* bih = (const float*)d_in[4];
  const float* bhh = (const float*)d_in[5];
  float* out = (float*)d_out;

  // workspace layout: x_proj f32 (96 MB) | W_ihT f16 | W_hhT f16
  char* ws = (char*)d_ws;
  float*    xproj = (float*)ws;
  _Float16* wihT  = (_Float16*)(ws + (size_t)T_LEN * B_SZ * G_SZ * 4);
  _Float16* whhT  = (_Float16*)((char*)wihT + (size_t)H_SZ * G_SZ * 2);

  wtrans_kernel<<<768, 256, 0, stream>>>(Wih, wihT);
  wtrans_kernel<<<768, 256, 0, stream>>>(Whh, whhT);

  dim3 g1(2048, 6);                                    // 2048 M-tiles x 48 N-tiles
  xproj_kernel<<<g1, 256, 0, stream>>>(inp, emb, wihT, bih, xproj);

  scan_kernel<<<4, 512, 0, stream>>>(xproj, whhT, bhh, out);
}
